// Attention_40913858462241
// MI455X (gfx1250) — matrix operand order repaired and checked by reading, not compiled, not run on hardware
//
#include <hip/hip_runtime.h>
#include <hip/hip_bf16.h>

// B=8, S=2048, E=1024, DK=64
#define BATCH 8
#define SEQ   2048
#define EDIM  1024
#define DKDIM 64

typedef __attribute__((ext_vector_type(16))) __bf16 v16bf;
typedef __attribute__((ext_vector_type(8)))  float  v8f;

union BF16x16 { v16bf v; __bf16 h[16]; };

__device__ __forceinline__ v8f wmma_bf16(v16bf a, v16bf b, v8f c) {
    // D = A(16x32 bf16) * B(32x16 bf16) + C(16x16 f32)
    return __builtin_amdgcn_wmma_f32_16x16x32_bf16(
        /*neg_a=*/false, a, /*neg_b=*/false, b,
        /*c_mod=*/(short)0, c, /*reuse_a=*/false, /*reuse_b=*/false);
}

// ---------------------------------------------------------------------------
// Kernel 1: fused Q/K/V linear projections (x @ W^T + b), f32 -> bf16.
//   task = (mat, mtile); one wave computes a full 16x64 output strip:
//   A operand (x rows) is loaded ONCE per k-step and reused across the 4
//   ntiles (4 WMMA accumulators) -> x is read exactly once from HBM
//   (192 MB total instead of 768 MB). W (256 KB each) stays L2-resident.
//   mat 0 -> Qb [B*S,64], mat 1 -> Kb [B*S,64], mat 2 -> Vt [B][64][S].
// ---------------------------------------------------------------------------
__global__ __launch_bounds__(256) void proj_qkv_kernel(
    const float* __restrict__ xq, const float* __restrict__ xk,
    const float* __restrict__ xv,
    const float* __restrict__ wq, const float* __restrict__ bq,
    const float* __restrict__ wk, const float* __restrict__ bk,
    const float* __restrict__ wv, const float* __restrict__ bv,
    __bf16* __restrict__ Qb, __bf16* __restrict__ Kb, __bf16* __restrict__ Vt)
{
    const int lane = threadIdx.x & 31;
    const int wave = threadIdx.x >> 5;
    const int ln   = lane & 15;   // A: row within tile / B,C: column within tile
    const int lg   = lane >> 4;   // half-wave group

    const int task  = blockIdx.x * 8 + wave;     // 0 .. 3071
    const int mat   = task >> 10;                // 0=Q 1=K 2=V
    const int mtile = task & 1023;               // 0 .. 1023 (B*S/16)

    const float* x; const float* w; const float* bias;
    if      (mat == 0) { x = xq; w = wq; bias = bq; }
    else if (mat == 1) { x = xk; w = wk; bias = bk; }
    else               { x = xv; w = wv; bias = bv; }

    const int m0 = mtile * 16;
    const long long rowA = (long long)(m0 + ln) * EDIM;   // A: lane = row m

    v8f cc[4] = {};                               // one accumulator per ntile
    for (int kk = 0; kk < EDIM / 32; ++kk) {
        BF16x16 a;
        const int e0 = kk * 32 + 8 * lg;          // A chunk0: K = 8g .. 8g+7
        const int e1 = kk * 32 + 16 + 8 * lg;     // A chunk1: K = 16+8g .. 23+8g
#pragma unroll
        for (int i = 0; i < 8; ++i)  a.h[i]     = (__bf16)x[rowA + e0 + i];
#pragma unroll
        for (int i = 0; i < 8; ++i)  a.h[8 + i] = (__bf16)x[rowA + e1 + i];
#pragma unroll
        for (int nt = 0; nt < 4; ++nt) {
            BF16x16 b;
            const long long rowB = (long long)(nt * 16 + ln) * EDIM;  // B: lane = col d
#pragma unroll
            for (int i = 0; i < 16; ++i)
                b.h[i] = (__bf16)w[rowB + kk * 32 + 8 * lg + (i & 7) + 16 * (i >> 3)];
            cc[nt] = wmma_bf16(a.v, b.v, cc[nt]);
        }
    }

#pragma unroll
    for (int nt = 0; nt < 4; ++nt) {
        const int d0 = nt * 16;
        const float bb = bias[d0 + ln];           // C: lane = column
        if (mat < 2) {
            __bf16* outp = (mat == 0) ? Qb : Kb;
#pragma unroll
            for (int r = 0; r < 8; ++r) {         // C row = r + 8*lg
                int row = m0 + r + 8 * lg;
                outp[(long long)row * DKDIM + d0 + ln] = (__bf16)(cc[nt][r] + bb);
            }
        } else {
            const int bidx = m0 >> 11;            // / SEQ
            const int s0   = m0 & (SEQ - 1);
#pragma unroll
            for (int r = 0; r < 8; ++r) {
                int s = s0 + r + 8 * lg;
                Vt[((long long)bidx * DKDIM + d0 + ln) * SEQ + s] = (__bf16)(cc[nt][r] + bb);
            }
        }
    }
}

// ---------------------------------------------------------------------------
// Kernel 2: fused scores + mask + softmax + weights@V for one (batch, 16-row)
// strip. 8 waves / 256 threads. 16x2048 f32 score strip lives in LDS (128 KB,
// fits the 320 KB WGP LDS with 2-WG occupancy). Phase 1 is bound by the
// 128 MB int32 mask stream -> prefetch the next tile's mask rows / K-tile.
// ---------------------------------------------------------------------------
__global__ __launch_bounds__(256) void attn_fused_kernel(
    const __bf16* __restrict__ Qb, const __bf16* __restrict__ Kb,
    const __bf16* __restrict__ Vt, const int* __restrict__ mask,
    float* __restrict__ out_att, float* __restrict__ out_w)
{
    __shared__ float s_s[16][SEQ];        // 128 KB score / weight strip
    __shared__ float s_part[8][16][16];   // 8 KB partial sums for phase 3

    const int bidx = blockIdx.x >> 7;     // / (SEQ/16)
    const int it   = blockIdx.x & 127;
    const int i0   = it * 16;
    const int lane = threadIdx.x & 31;
    const int wave = threadIdx.x >> 5;
    const int ln   = lane & 15;
    const int lg   = lane >> 4;
    const float scale = 0.125f;           // 1/sqrt(64)

    // ---- Phase 1: scores = scale * Q K^T, with mask fused --------------------
    // A operand (Q rows i0..i0+15, DK=64 -> two k-steps) loaded once per wave.
    BF16x16 a0, a1;
    {
        const __bf16* qrow = Qb + ((long long)bidx * SEQ + i0 + ln) * DKDIM;
#pragma unroll
        for (int i = 0; i < 8; ++i) { a0.h[i]     = qrow[8 * lg + i];
                                      a0.h[8 + i] = qrow[16 + 8 * lg + i];
                                      a1.h[i]     = qrow[32 + 8 * lg + i];
                                      a1.h[8 + i] = qrow[48 + 8 * lg + i]; }
    }
    const long long mrow0 = ((long long)bidx * SEQ + i0) * SEQ;   // mask strip base
    for (int t = wave; t < SEQ / 16; t += 8) {
        const int j0 = t * 16;
        // Prefetch next iteration's mask rows (HBM stream) and K tile (L2).
        if (t + 8 < SEQ / 16) {
            const int jn = (t + 8) * 16;
            __builtin_prefetch(&mask[mrow0 + (lg * 8) * SEQ + jn + ln * 4], 0, 1);
            __builtin_prefetch(&Kb[((long long)bidx * SEQ + jn + ln) * DKDIM], 0, 3);
        }
        BF16x16 b0, b1;
        const __bf16* krow = Kb + ((long long)bidx * SEQ + j0 + ln) * DKDIM;
#pragma unroll
        for (int i = 0; i < 16; ++i) { b0.h[i] = krow[8 * lg + (i & 7) + 16 * (i >> 3)];
                                       b1.h[i] = krow[32 + 8 * lg + (i & 7) + 16 * (i >> 3)]; }
        v8f c = {};
        c = wmma_bf16(a0.v, b0.v, c);
        c = wmma_bf16(a1.v, b1.v, c);
#pragma unroll
        for (int r = 0; r < 8; ++r) {
            const int m = r + 8 * lg;
            const int j = j0 + ln;
            const int mk = mask[mrow0 + (long long)m * SEQ + j];
            s_s[m][j] = (mk == 0) ? -1e9f : c[r] * scale;
        }
    }
    __syncthreads();

    // ---- Phase 2: row softmax (2 rows per wave), stream weights to d_out -----
#pragma unroll
    for (int rr = 0; rr < 2; ++rr) {
        const int m = wave * 2 + rr;
        float mx = -1e30f;
        for (int j = lane; j < SEQ; j += 32) mx = fmaxf(mx, s_s[m][j]);
#pragma unroll
        for (int off = 16; off > 0; off >>= 1) mx = fmaxf(mx, __shfl_xor(mx, off, 32));
        float sum = 0.f;
        for (int j = lane; j < SEQ; j += 32) {
            const float e = __expf(s_s[m][j] - mx);
            s_s[m][j] = e;
            sum += e;
        }
#pragma unroll
        for (int off = 16; off > 0; off >>= 1) sum += __shfl_xor(sum, off, 32);
        const float inv = 1.0f / sum;
        float* wout = out_w + ((long long)bidx * SEQ + i0 + m) * SEQ;
        for (int j = lane; j < SEQ; j += 32) {
            const float wv = s_s[m][j] * inv;
            s_s[m][j] = wv;
            wout[j]   = wv;
        }
    }
    __syncthreads();

    // ---- Phase 3: out = weights @ V. 4 d-tiles x 2 K-halves across 8 waves ---
    const int tile  = wave >> 1;          // d0 = 16*tile
    const int khalf = wave & 1;           // j in [khalf*1024, +1024)
    const int jbase = khalf * (SEQ / 2);
    v8f c = {};
    const __bf16* vrow = Vt + ((long long)bidx * DKDIM + tile * 16 + ln) * SEQ + jbase;
    for (int kk = 0; kk < (SEQ / 2) / 32; ++kk) {
        const int jb = kk * 32;
        BF16x16 a, b;
#pragma unroll
        for (int i = 0; i < 8; ++i) {
            a.h[i]     = (__bf16)s_s[ln][jbase + jb + 8 * lg + i];
            a.h[8 + i] = (__bf16)s_s[ln][jbase + jb + 16 + 8 * lg + i];
        }
#pragma unroll
        for (int i = 0; i < 16; ++i) b.h[i] = vrow[jb + 8 * lg + (i & 7) + 16 * (i >> 3)];
        c = wmma_bf16(a.v, b.v, c);
    }
#pragma unroll
    for (int r = 0; r < 8; ++r) s_part[wave][r + 8 * lg][ln] = c[r];
    __syncthreads();

    for (int e = threadIdx.x; e < 16 * DKDIM; e += 256) {
        const int m = e >> 6, d = e & 63;
        const int tl = d >> 4, n = d & 15;
        const float val = s_part[2 * tl][m][n] + s_part[2 * tl + 1][m][n];
        out_att[((long long)bidx * SEQ + i0 + m) * DKDIM + d] = val;
    }
}

// ---------------------------------------------------------------------------
extern "C" void kernel_launch(void* const* d_in, const int* in_sizes, int n_in,
                              void* d_out, int out_size, void* d_ws, size_t ws_size,
                              hipStream_t stream) {
    const float* k_in = (const float*)d_in[0];
    const float* q_in = (const float*)d_in[1];
    const float* v_in = (const float*)d_in[2];
    const int*   mask = (const int*)d_in[3];
    const float* w_q  = (const float*)d_in[4];
    const float* b_q  = (const float*)d_in[5];
    const float* w_k  = (const float*)d_in[6];
    const float* b_k  = (const float*)d_in[7];
    const float* w_v  = (const float*)d_in[8];
    const float* b_v  = (const float*)d_in[9];

    const size_t nqkv = (size_t)BATCH * SEQ * DKDIM;
    __bf16* Qb = (__bf16*)d_ws;
    __bf16* Kb = Qb + nqkv;
    __bf16* Vt = Kb + nqkv;          // [B][DK][S] transposed

    float* out_att = (float*)d_out;                  // [B,S,DK]
    float* out_w   = out_att + nqkv;                 // [B,S,S]

    // 3 mats * 1024 m-tiles = 3072 wave tasks / 8 waves per block
    proj_qkv_kernel<<<384, 256, 0, stream>>>(q_in, k_in, v_in,
                                             w_q, b_q, w_k, b_k, w_v, b_v,
                                             Qb, Kb, Vt);
    // one block per (batch, 16-row strip): 8 * 128 = 1024 blocks
    attn_fused_kernel<<<BATCH * (SEQ / 16), 256, 0, stream>>>(Qb, Kb, Vt, mask,
                                                              out_att, out_w);
}
